// GroupQueryAttention_79276506350263
// MI455X (gfx1250) — matrix operand order repaired and checked by reading, not compiled, not run on hardware
//
#include <hip/hip_runtime.h>
#include <cstddef>
#include <cstdint>

typedef unsigned short u16;
typedef __attribute__((ext_vector_type(16))) __bf16 v16bf;
typedef __attribute__((ext_vector_type(8)))  float  v8f;

#define BATCH  2
#define SEQ    1024
#define CMODEL 4096
#define NQH    32
#define NKVH   8
#define HDIM   128
#define BT_    (BATCH*SEQ)      // 2048
#define HD_    (NQH*HDIM)       // 4096
#define GD_    (NKVH*HDIM)      // 1024

static __device__ __forceinline__ u16 f2bf(float f) {
  unsigned u = __float_as_uint(f);
  u += 0x7FFFu + ((u >> 16) & 1u);   // round to nearest even
  return (u16)(u >> 16);
}

static __device__ __forceinline__ v8f vzero8() {
  v8f z;
#pragma unroll
  for (int i = 0; i < 8; ++i) z[i] = 0.0f;
  return z;
}

static __device__ __forceinline__ v8f wmma_bf16(v16bf a, v16bf b, v8f c) {
  return __builtin_amdgcn_wmma_f32_16x16x32_bf16(false, a, false, b, (short)0, c,
                                                 false, false);
}

// Async global->LDS 16B copy (CDNA5 GLOBAL_LOAD_ASYNC_TO_LDS_B128, ASYNCcnt).
// LDS destination address = low 32 bits of the generic pointer.
static __device__ __forceinline__ void g2l_b128(void* ldsDst, const void* gSrc) {
  asm volatile("global_load_async_to_lds_b128 %0, %1, off"
               :
               : "v"((unsigned)(size_t)ldsDst), "v"(gSrc)
               : "memory");
}
static __device__ __forceinline__ void wait_async0() {
  asm volatile("s_wait_asynccnt 0x0" ::: "memory");
}

// ---------------- f32 -> bf16 elementwise ----------------
__global__ void k_cvt_bf16(const float* __restrict__ in, u16* __restrict__ out, int n) {
  int i = blockIdx.x * blockDim.x + threadIdx.x;
  if (i < n) out[i] = f2bf(in[i]);
}

// ---------------- transpose f32 [K][N] -> bf16 [N][K] ----------------
__global__ void k_transpose_bf16(const float* __restrict__ W, u16* __restrict__ Wt,
                                 int K, int N) {
  __shared__ float tile[32][33];
  int k0 = blockIdx.y * 32, n0 = blockIdx.x * 32;
  int tx = threadIdx.x, ty = threadIdx.y;
  for (int i = ty; i < 32; i += 8)
    tile[i][tx] = W[(size_t)(k0 + i) * N + (n0 + tx)];
  __syncthreads();
  for (int i = ty; i < 32; i += 8)
    Wt[(size_t)(n0 + i) * K + (k0 + tx)] = f2bf(tile[tx][i]);
}

// ---------------- bf16 WMMA GEMM: C[M][N] = A[M][K] * Bt[N][K]^T (+bias) ----
// Double-buffered LDS, async global->LDS DMA, one barrier per K-step.
__global__ __launch_bounds__(256) void k_gemm_bf16(
    const u16* __restrict__ A, const u16* __restrict__ Bt,
    float* __restrict__ C, const float* __restrict__ bias,
    int M, int N, int K) {
  __shared__ u16 lA[2][128 * 32];
  __shared__ u16 lB[2][128 * 32];
  const int tid = threadIdx.x;
  const int wave = tid >> 5, lane = tid & 31;
  const int h = lane >> 4, ln = lane & 15;
  const int wm = (wave >> 1) * 32;   // 4 waves along M
  const int wn = (wave & 1) * 64;    // 2 waves along N
  const int bm = blockIdx.y * 128, bn = blockIdx.x * 128;

  // per-thread chunk coordinates (512 x 16B chunks per 128x32 bf16 tile)
  const int i0 = tid, i1 = tid + 256;
  const int r0 = i0 >> 2, c0 = i0 & 3;
  const int r1 = i1 >> 2, c1 = i1 & 3;

  auto issue = [&](int buf, int k0) {
    g2l_b128(&lA[buf][i0 * 8], A + (size_t)(bm + r0) * K + k0 + c0 * 8);
    g2l_b128(&lB[buf][i0 * 8], Bt + (size_t)(bn + r0) * K + k0 + c0 * 8);
    g2l_b128(&lA[buf][i1 * 8], A + (size_t)(bm + r1) * K + k0 + c1 * 8);
    g2l_b128(&lB[buf][i1 * 8], Bt + (size_t)(bn + r1) * K + k0 + c1 * 8);
  };

  v8f acc[2][4];
#pragma unroll
  for (int i = 0; i < 2; ++i)
#pragma unroll
    for (int j = 0; j < 4; ++j) acc[i][j] = vzero8();

  const int nk = K >> 5;
  issue(0, 0);
  for (int kk = 0; kk < nk; ++kk) {
    wait_async0();          // this wave's async stores for tile kk have landed
    __syncthreads();        // all waves' tile kk visible; tile kk-1 compute done
    if (kk + 1 < nk) issue((kk + 1) & 1, (kk + 1) * 32);

    const u16* sA = lA[kk & 1];
    const u16* sB = lB[kk & 1];
    v16bf af[2], bfr[4];
#pragma unroll
    for (int i = 0; i < 2; ++i) {                    // A-fragment layout
      uint4* p = (uint4*)&af[i];
      const u16* src = sA + (wm + i * 16 + ln) * 32;
      p[0] = *(const uint4*)(src + h * 8);
      p[1] = *(const uint4*)(src + 16 + h * 8);
    }
#pragma unroll
    for (int j = 0; j < 4; ++j) {                    // B-fragment layout
      uint4* p = (uint4*)&bfr[j];
      const u16* src = sB + (wn + j * 16 + ln) * 32;
      p[0] = *(const uint4*)(src + h * 8);
      p[1] = *(const uint4*)(src + 16 + h * 8);
    }
#pragma unroll
    for (int i = 0; i < 2; ++i)
#pragma unroll
      for (int j = 0; j < 4; ++j)
        acc[i][j] = wmma_bf16(af[i], bfr[j], acc[i][j]);
  }

#pragma unroll
  for (int i = 0; i < 2; ++i)
#pragma unroll
    for (int j = 0; j < 4; ++j) {
      int col = bn + wn + j * 16 + ln;
      float bv = bias ? bias[col] : 0.0f;
      int row0 = bm + wm + i * 16 + 8 * h;           // C-layout: m = e + 8h
#pragma unroll
      for (int e = 0; e < 8; ++e)
        C[(size_t)(row0 + e) * N + col] = acc[i][j][e] + bv;
    }
}

// ---------------- RoPE + repack: f32 [B*T][Hn*128] -> bf16 [B][Hn][T][128] --
__global__ void k_rope_pack(const float* __restrict__ Xf, u16* __restrict__ Xc,
                            int Hn, int n) {
  int idx = blockIdx.x * blockDim.x + threadIdx.x;
  if (idx >= n) return;
  int i = idx & 63;          // rotary pair index
  int r = idx >> 6;
  int head = r % Hn;
  int bt = r / Hn;
  int t = bt & (SEQ - 1);
  int b = bt >> 10;
  // freq = theta^{-(2i)/128}; ln(10000)/64 = 0.1439115683
  float ang = (float)t * __expf(-(float)i * 0.1439115683f);
  float sn, cs;
  __sincosf(ang, &sn, &cs);
  const float* src = Xf + (size_t)bt * (Hn * HDIM) + head * HDIM + 2 * i;
  float xr = src[0], xi = src[1];
  size_t ob = (((size_t)b * Hn + head) * SEQ + t) * HDIM + 2 * i;
  Xc[ob]     = f2bf(xr * cs - xi * sn);
  Xc[ob + 1] = f2bf(xr * sn + xi * cs);
}

// ---------------- V repack transposed: f32 [B*T][G*128] -> bf16 [B][G][128][T]
__global__ void k_pack_vT(const float* __restrict__ Vf, u16* __restrict__ Vt, int n) {
  int idx = blockIdx.x * blockDim.x + threadIdx.x;
  if (idx >= n) return;
  int t  = idx & (SEQ - 1);
  int r  = idx >> 10;
  int d  = r & (HDIM - 1);
  int r2 = r >> 7;
  int g  = r2 & (NKVH - 1);
  int b  = r2 >> 3;
  Vt[idx] = f2bf(Vf[((size_t)(b * SEQ + t)) * GD_ + g * HDIM + d]);
}

// ---------------- causal flash attention (bf16 WMMA, f32 accum) -------------
__global__ __launch_bounds__(128) void k_flash_attn(
    const u16* __restrict__ Qc,   // [B][H][T][128]
    const u16* __restrict__ Kc,   // [B][G][T][128]
    const u16* __restrict__ Vt,   // [B][G][128][T]
    u16* __restrict__ AO) {       // [B*T][H*128] bf16
  __shared__ u16 lK[32 * 128];    // [key][d]
  __shared__ u16 lV[128 * 32];    // [d][key]
  __shared__ u16 lP[4 * 16 * 32]; // per-wave P exchange

  const int qt = blockIdx.x, head = blockIdx.y, b = blockIdx.z;
  const int g = head >> 2;        // H/G = 4
  const int tid = threadIdx.x;
  const int wave = tid >> 5, lane = tid & 31;
  const int h = lane >> 4, ln = lane & 15;
  const int qbase = qt * 64 + wave * 16;

  const u16* Qp = Qc + ((size_t)(b * NQH + head)) * SEQ * HDIM;
  const u16* Kp = Kc + ((size_t)(b * NKVH + g)) * SEQ * HDIM;
  const u16* Vp = Vt + ((size_t)(b * NKVH + g)) * HDIM * SEQ;

  v16bf qf[4];
#pragma unroll
  for (int dd = 0; dd < 4; ++dd) {                 // Q rows as A-fragments
    uint4* p = (uint4*)&qf[dd];
    const u16* src = Qp + (size_t)(qbase + ln) * HDIM + dd * 32;
    p[0] = *(const uint4*)(src + h * 8);
    p[1] = *(const uint4*)(src + 16 + h * 8);
  }

  v8f o[8];
#pragma unroll
  for (int d = 0; d < 8; ++d) o[d] = vzero8();
  float mi[8], li[8];
#pragma unroll
  for (int j = 0; j < 8; ++j) { mi[j] = -1e30f; li[j] = 0.0f; }

  const int ktEnd = (qt * 64 + 63) >> 5;           // uniform over block
  for (int kt = 0; kt <= ktEnd; ++kt) {
    const int kb = kt * 32;
#pragma unroll
    for (int q = 0; q < 4; ++q) {                  // K tile 32x128, async DMA
      int i = tid + q * 128;
      int r = i >> 4, c = i & 15;
      g2l_b128(lK + i * 8, Kp + (size_t)(kb + r) * HDIM + c * 8);
    }
#pragma unroll
    for (int q = 0; q < 4; ++q) {                  // V tile 128x32, async DMA
      int i = tid + q * 128;
      int r = i >> 2, c = i & 3;
      g2l_b128(lV + i * 8, Vp + (size_t)r * SEQ + kb + c * 8);
    }
    wait_async0();
    __syncthreads();

    v8f s[2];
    s[0] = vzero8(); s[1] = vzero8();
#pragma unroll
    for (int nt = 0; nt < 2; ++nt) {               // S = Q K^T over d-chunks
#pragma unroll
      for (int dd = 0; dd < 4; ++dd) {
        v16bf kf;
        uint4* p = (uint4*)&kf;
        const u16* src = lK + (nt * 16 + ln) * HDIM + dd * 32;
        p[0] = *(const uint4*)(src + h * 8);
        p[1] = *(const uint4*)(src + 16 + h * 8);
        s[nt] = wmma_bf16(qf[dd], kf, s[nt]);
      }
    }

    const float scale = 0.08838834764831843f;      // 1/sqrt(128)
    u16* myP = lP + wave * 512;
#pragma unroll
    for (int j = 0; j < 8; ++j) {                  // online softmax per row
      const int qrow = qbase + j + 8 * h;
      float s0 = s[0][j] * scale;
      float s1 = s[1][j] * scale;
      if (kb + ln > qrow)      s0 = -1e30f;        // causal mask
      if (kb + 16 + ln > qrow) s1 = -1e30f;
      float mx = fmaxf(s0, s1);
#pragma unroll
      for (int off = 1; off < 16; off <<= 1) mx = fmaxf(mx, __shfl_xor(mx, off, 32));
      const float mnew  = fmaxf(mi[j], mx);
      const float alpha = __expf(mi[j] - mnew);
      const float p0 = __expf(s0 - mnew);
      const float p1 = __expf(s1 - mnew);
      float rs = p0 + p1;
#pragma unroll
      for (int off = 1; off < 16; off <<= 1) rs += __shfl_xor(rs, off, 32);
      li[j] = li[j] * alpha + rs;
      mi[j] = mnew;
#pragma unroll
      for (int d = 0; d < 8; ++d) o[d][j] *= alpha;
      const int m = j + 8 * h;                     // C-layout -> LDS
      myP[m * 32 + ln]      = f2bf(p0);
      myP[m * 32 + 16 + ln] = f2bf(p1);
    }

    v16bf pf;                                      // reload P in A-layout
    {
      uint4* p = (uint4*)&pf;
      const u16* src = myP + ln * 32;
      p[0] = *(const uint4*)(src + h * 8);
      p[1] = *(const uint4*)(src + 16 + h * 8);
    }
#pragma unroll
    for (int dt = 0; dt < 8; ++dt) {               // O += P @ V
      v16bf vf;
      uint4* p = (uint4*)&vf;
      const u16* src = lV + (dt * 16 + ln) * 32;
      p[0] = *(const uint4*)(src + h * 8);
      p[1] = *(const uint4*)(src + 16 + h * 8);
      o[dt] = wmma_bf16(pf, vf, o[dt]);
    }
    __syncthreads();   // all waves done with lK/lV before next async overwrite
  }

#pragma unroll
  for (int dt = 0; dt < 8; ++dt)
#pragma unroll
    for (int j = 0; j < 8; ++j) {
      const int t   = qbase + j + 8 * h;
      const int col = head * HDIM + dt * 16 + ln;
      AO[((size_t)(b * SEQ + t)) * HD_ + col] = f2bf(o[dt][j] / li[j]);
    }
}

// ---------------------------------------------------------------------------
extern "C" void kernel_launch(void* const* d_in, const int* in_sizes, int n_in,
                              void* d_out, int out_size, void* d_ws, size_t ws_size,
                              hipStream_t stream) {
  const float* x  = (const float*)d_in[0];
  const float* Wq = (const float*)d_in[1];
  const float* Wk = (const float*)d_in[2];
  const float* Wv = (const float*)d_in[3];
  const float* Wo = (const float*)d_in[4];
  const float* bo = (const float*)d_in[5];
  float* out = (float*)d_out;
  (void)in_sizes; (void)n_in; (void)out_size; (void)ws_size;

  char* ws = (char*)d_ws;
  size_t off = 0;
  auto alloc = [&](size_t bytes) -> void* {
    void* p = ws + off;
    off += (bytes + 255) & ~(size_t)255;
    return p;
  };

  u16*   xb  = (u16*)  alloc((size_t)BT_ * CMODEL * 2);
  u16*   WqT = (u16*)  alloc((size_t)HD_ * CMODEL * 2);
  u16*   WkT = (u16*)  alloc((size_t)GD_ * CMODEL * 2);
  u16*   WvT = (u16*)  alloc((size_t)GD_ * CMODEL * 2);
  u16*   WoT = (u16*)  alloc((size_t)CMODEL * HD_ * 2);
  float* Qf  = (float*)alloc((size_t)BT_ * HD_ * 4);
  float* Kf  = (float*)alloc((size_t)BT_ * GD_ * 4);
  float* Vf  = (float*)alloc((size_t)BT_ * GD_ * 4);
  u16*   Qc  = (u16*)  alloc((size_t)BT_ * HD_ * 2);
  u16*   Kc  = (u16*)  alloc((size_t)BT_ * GD_ * 2);
  u16*   Vt  = (u16*)  alloc((size_t)BT_ * GD_ * 2);
  u16*   AO  = (u16*)  alloc((size_t)BT_ * HD_ * 2);

  { int n = BT_ * CMODEL;
    k_cvt_bf16<<<n / 256, 256, 0, stream>>>(x, xb, n); }

  k_transpose_bf16<<<dim3(HD_ / 32, CMODEL / 32), dim3(32, 8), 0, stream>>>(Wq, WqT, CMODEL, HD_);
  k_transpose_bf16<<<dim3(GD_ / 32, CMODEL / 32), dim3(32, 8), 0, stream>>>(Wk, WkT, CMODEL, GD_);
  k_transpose_bf16<<<dim3(GD_ / 32, CMODEL / 32), dim3(32, 8), 0, stream>>>(Wv, WvT, CMODEL, GD_);
  k_transpose_bf16<<<dim3(CMODEL / 32, HD_ / 32), dim3(32, 8), 0, stream>>>(Wo, WoT, HD_, CMODEL);

  k_gemm_bf16<<<dim3(HD_ / 128, BT_ / 128), 256, 0, stream>>>(xb, WqT, Qf, nullptr, BT_, HD_, CMODEL);
  k_gemm_bf16<<<dim3(GD_ / 128, BT_ / 128), 256, 0, stream>>>(xb, WkT, Kf, nullptr, BT_, GD_, CMODEL);
  k_gemm_bf16<<<dim3(GD_ / 128, BT_ / 128), 256, 0, stream>>>(xb, WvT, Vf, nullptr, BT_, GD_, CMODEL);

  { int n = BT_ * NQH * 64;
    k_rope_pack<<<n / 256, 256, 0, stream>>>(Qf, Qc, NQH, n); }
  { int n = BT_ * NKVH * 64;
    k_rope_pack<<<n / 256, 256, 0, stream>>>(Kf, Kc, NKVH, n); }
  { int n = BT_ * GD_;
    k_pack_vT<<<n / 256, 256, 0, stream>>>(Vf, Vt, n); }

  k_flash_attn<<<dim3(SEQ / 64, NQH, BATCH), 128, 0, stream>>>(Qc, Kc, Vt, AO);

  k_gemm_bf16<<<dim3(CMODEL / 128, BT_ / 128), 256, 0, stream>>>(AO, WoT, out, bo, BT_, CMODEL, CMODEL);
}
